// Decoder_60730837565923
// MI455X (gfx1250) — matrix operand order repaired and checked by reading, not compiled, not run on hardware
//
#include <hip/hip_runtime.h>

typedef _Float16 h2  __attribute__((ext_vector_type(2)));
typedef _Float16 h8  __attribute__((ext_vector_type(8)));
typedef _Float16 h16 __attribute__((ext_vector_type(16)));
typedef float    f8  __attribute__((ext_vector_type(8)));

struct MlpParams {
    const float* W[9];
    const float* B[9];
    const float* x;   // [N,2] f32
    float*       y;   // [N]   f32
    int          N;
};

// Branchless softplus(beta=100)/beta, exact w.r.t. the reference's
// where(100x > 20, x, log1p(exp(100x))/100):
//   sp = log2(1 + exp2(min(144.2695*x, 28.8539))) * (ln2/100)
// saturates at 0.2 for x > 0.2, and softplus >= identity, so fmax(x, sp)
// selects the right branch with no compare/cndmask/exec divergence.
__device__ __forceinline__ float softplus100(float x) {
    const float a  = __builtin_amdgcn_exp2f(fminf(x * 144.26950408889634f,
                                                  28.853900817779268f));
    const float sp = __builtin_amdgcn_logf(1.0f + a) * 0.006931471805599453f;
    return fmaxf(x, sp);
}

// One wave (32 lanes) owns a 16-row batch tile. 8 waves/block share per-layer
// weights staged in LDS as zero-padded 112x128 f16 [N][K] row-major.
// GEMM: D(16x16,f32) += A(16x32,f16 acts) x B(32x16,f16 = W^T slice).
__global__ __launch_bounds__(256, 1)
void mlp9_wmma_f16(MlpParams p) {
    __shared__ __align__(16) _Float16 wlds[112 * 128];  // 28 KB weights (padded)
    __shared__ float blds[112];                          // bias
    __shared__ __align__(16) _Float16 act[8][16 * 128];  // 4 KB per-wave acts

    const int tid  = threadIdx.x;
    const int w    = tid >> 5;
    const int lane = tid & 31;
    const int l16  = lane & 15;
    const int hi   = lane >> 4;
    _Float16* A = act[w];

    // Zero this wave's activation tile (padding K>=dim must be 0 for WMMA).
    h8 z8 = {};
    #pragma unroll
    for (int i = 0; i < 8; ++i) ((h8*)A)[lane + 32 * i] = z8;

    const int rowBase = blockIdx.x * 128 + w * 16;

    // Load input (2 features) into K=0..1; keep f16 copy for the skip concat.
    h2 inPk = {};
    if (lane < 16) {
        const int r = rowBase + l16;
        float vx = 0.f, vy = 0.f;
        if (r < p.N) { const float2 t = ((const float2*)p.x)[r]; vx = t.x; vy = t.y; }
        inPk[0] = (_Float16)vx;
        inPk[1] = (_Float16)vy;
        *(h2*)&A[l16 * 128] = inPk;
    }

    const int inDim[9]  = {2, 100, 100, 100, 100, 100, 100, 100, 100};
    const int outDim[9] = {100, 100, 100, 98, 100, 100, 100, 100, 1};

    #pragma unroll
    for (int l = 0; l < 9; ++l) {
        const int in  = inDim[l];
        const int out = outDim[l];

        __syncthreads();  // prior layer done reading wlds before we overwrite
        {
            const float* __restrict__ Wg = p.W[l];
            // 112*128 f16 = 7168 dword slots = 256 threads x 28
            #pragma unroll 1
            for (int i = 0; i < 28; ++i) {
                const int slot = tid + i * 256;
                const int n  = slot >> 6;         // padded out-row
                const int kk = (slot & 63) << 1;  // padded in-col (pair)
                h2 pk;
                pk[0] = (_Float16)((n < out && kk     < in) ? Wg[n * in + kk]     : 0.f);
                pk[1] = (_Float16)((n < out && kk + 1 < in) ? Wg[n * in + kk + 1] : 0.f);
                *(h2*)&wlds[slot * 2] = pk;
            }
            if (tid < 112) blds[tid] = (tid < out) ? p.B[l][tid] : 0.f;
        }
        __syncthreads();

        // A-fragments (16x32 f16): lanes 0-15 K=0..7 / 16..23, lanes 16-31 K=8..15 / 24..31
        const int kf = (in + 31) >> 5;
        h16 a[4];
        #pragma unroll
        for (int f = 0; f < 4; ++f) {
            if (f < kf) {
                const _Float16* ap = &A[l16 * 128 + hi * 8 + f * 32];
                ((h8*)&a[f])[0] = *(const h8*)ap;
                ((h8*)&a[f])[1] = *(const h8*)(ap + 16);
            }
        }

        const int nt = (out + 15) >> 4;
        #pragma unroll
        for (int n = 0; n < nt; ++n) {
            f8 acc = {};
            #pragma unroll
            for (int f = 0; f < kf; ++f) {
                // B-fragment (32x16 f16): lane = col; lanes 0-15 K=0..7 / 16..23, lanes 16-31 K=8..15 / 24..31
                const _Float16* bp = &wlds[(n * 16 + l16) * 128 + hi * 8 + f * 32];
                h16 bm;
                ((h8*)&bm)[0] = *(const h8*)bp;
                ((h8*)&bm)[1] = *(const h8*)(bp + 16);
                acc = __builtin_amdgcn_wmma_f32_16x16x32_f16(
                    false, a[f], false, bm, (short)0, acc, false, false);
            }

            const int col  = n * 16 + l16;       // D layout: col = lane%16
            const float bi = blds[col];
            #pragma unroll
            for (int j = 0; j < 8; ++j) acc[j] += bi;

            if (l < 8) {
                #pragma unroll
                for (int j = 0; j < 8; ++j) acc[j] = softplus100(acc[j]);
                if (col < out) {
                    const int koff = (l == 3) ? 2 : 0;   // layer-3 output sits at K=2..99
                    #pragma unroll
                    for (int j = 0; j < 8; ++j)          // D row = j + 8*hi
                        A[(j + hi * 8) * 128 + koff + col] = (_Float16)acc[j];
                }
            } else {
                if (col == 0) {                          // single output feature
                    #pragma unroll
                    for (int j = 0; j < 8; ++j) {
                        const int r = rowBase + hi * 8 + j;
                        if (r < p.N) p.y[r] = acc[j];
                    }
                }
            }
        }

        // Skip connection: restore original 2 input features at K=0..1
        if (l == 3 && lane < 16) *(h2*)&A[l16 * 128] = inPk;
    }
}

extern "C" void kernel_launch(void* const* d_in, const int* in_sizes, int n_in,
                              void* d_out, int out_size, void* d_ws, size_t ws_size,
                              hipStream_t stream) {
    (void)n_in; (void)out_size; (void)d_ws; (void)ws_size;
    MlpParams p;
    p.x = (const float*)d_in[0];
    for (int l = 0; l < 9; ++l) {
        p.W[l] = (const float*)d_in[1 + 2 * l];
        p.B[l] = (const float*)d_in[2 + 2 * l];
    }
    p.y = (float*)d_out;
    p.N = in_sizes[0] / 2;
    const int grid = (p.N + 127) / 128;
    hipLaunchKernelGGL(mlp9_wmma_f16, dim3(grid), dim3(256), 0, stream, p);
}
